// MixtureOfExperts_68539088109739
// MI455X (gfx1250) — compile-verified, matrix operand order checked by reading, not run on hardware
//
#include <hip/hip_runtime.h>

// MoE: B=2,S=2048,D=1024,E=8,F=4096,K=2 (top-2 routing)
#define Bm   2
#define Sm   2048
#define Dm   1024
#define Em   8
#define Fm   4096
#define BSm  (Bm*Sm)           // 4096 tokens
#define TILE_M 32              // tokens per workgroup tile (2 WMMA row-tiles)
#define MAX_TILES 128          // ceil(BSm / TILE_M)
#define FBLK 128
#define XPAD 8                 // ushort pad -> row stride 1032 (kills LDS bank conflicts)
#define HPAD 8

typedef __attribute__((ext_vector_type(16))) __bf16          v16bf;
typedef __attribute__((ext_vector_type(8)))  float           v8f;
typedef __attribute__((ext_vector_type(8)))  unsigned short  ushort8;

union Frag { ushort8 u8[2]; v16bf v; };

static __device__ __forceinline__ unsigned short f2bf(float f) {
  unsigned int u = __float_as_uint(f);
  unsigned int r = u + 0x7FFFu + ((u >> 16) & 1u);   // round-to-nearest-even
  return (unsigned short)(r >> 16);
}

// ---------------------------------------------------------------- zero counters
__global__ void moe_zero_cnt(int* __restrict__ cnt) {
  if (threadIdx.x < Em) cnt[threadIdx.x] = 0;
}

// ------------------------------------------------- w1 [E,D,F] f32 -> [E,F,D] bf16
__global__ void moe_conv_w1(const float* __restrict__ w1, unsigned short* __restrict__ w1bf) {
  size_t idx = (size_t)blockIdx.x * blockDim.x + threadIdx.x;
  if (idx >= (size_t)Em * Dm * Fm) return;
  size_t e   = idx / ((size_t)Dm * Fm);
  size_t rem = idx - e * (size_t)Dm * Fm;
  size_t d   = rem / Fm;
  size_t f   = rem - d * Fm;
  w1bf[(e * Fm + f) * Dm + d] = f2bf(w1[idx]);
}

// ------------------------------------------------- w2 [E,F,D] f32 -> [E,D,F] bf16
__global__ void moe_conv_w2(const float* __restrict__ w2, unsigned short* __restrict__ w2bf) {
  size_t idx = (size_t)blockIdx.x * blockDim.x + threadIdx.x;
  if (idx >= (size_t)Em * Fm * Dm) return;
  size_t e   = idx / ((size_t)Fm * Dm);
  size_t rem = idx - e * (size_t)Fm * Dm;
  size_t f   = rem / Dm;
  size_t d   = rem - f * Dm;
  w2bf[(e * Dm + d) * Fm + f] = f2bf(w2[idx]);
}

// ------------------------------------------------------------------- router
__global__ void moe_router(const float* __restrict__ x,
                           const float* __restrict__ rw,
                           const float* __restrict__ rb,
                           int*   __restrict__ cnt,
                           int*   __restrict__ lists,
                           float* __restrict__ gates) {
  const int wavesPerBlk = blockDim.x >> 5;
  const int token = blockIdx.x * wavesPerBlk + (threadIdx.x >> 5);
  const int lane  = threadIdx.x & 31;
  if (token >= BSm) return;

  float acc[Em];
  #pragma unroll
  for (int e = 0; e < Em; e++) acc[e] = 0.0f;

  const float* xr = x + (size_t)token * Dm;
  for (int d = lane; d < Dm; d += 32) {
    float xv = xr[d];
    #pragma unroll
    for (int e = 0; e < Em; e++) acc[e] += xv * rw[d * Em + e];
  }
  #pragma unroll
  for (int e = 0; e < Em; e++) {
    #pragma unroll
    for (int off = 16; off > 0; off >>= 1)
      acc[e] += __shfl_xor(acc[e], off, 32);
    acc[e] += rb[e];
  }
  float mx = acc[0];
  #pragma unroll
  for (int e = 1; e < Em; e++) mx = fmaxf(mx, acc[e]);
  float p[Em], tot = 0.0f;
  #pragma unroll
  for (int e = 0; e < Em; e++) { p[e] = expf(acc[e] - mx); tot += p[e]; }
  float b0 = -1.0f, b1v = -1.0f; int i0 = 0, i1 = 0;
  #pragma unroll
  for (int e = 0; e < Em; e++) {
    if (p[e] > b0)       { b1v = b0; i1 = i0; b0 = p[e]; i0 = e; }
    else if (p[e] > b1v) { b1v = p[e]; i1 = e; }
  }
  float pr0 = b0 / tot, pr1 = b1v / tot;
  float den = pr0 + pr1 + 1e-8f;
  float g0 = pr0 / den, g1 = pr1 / den;

  if (lane == 0) {
    int q0 = atomicAdd(&cnt[i0], 1);
    lists[i0 * BSm + q0] = (token << 1) | 0;
    gates[i0 * BSm + q0] = g0;
    int q1 = atomicAdd(&cnt[i1], 1);
    lists[i1 * BSm + q1] = (token << 1) | 1;
    gates[i1 * BSm + q1] = g1;
  }
}

// ------------------------------------------------------------------- fused FFN
// block = 8 waves, one (expert, 32-token tile). Each B fragment (w1/w2 slice
// from L2) feeds TWO WMMAs (row tiles m=0..15 and m=16..31) -> halves L2
// weight traffic per FLOP vs a 16-token tile.
__global__ __launch_bounds__(256) void moe_ffn(
    const unsigned short* __restrict__ w1bf,   // [E][F][D] bf16
    const unsigned short* __restrict__ w2bf,   // [E][D][F] bf16
    const float* __restrict__ b1,              // [E][F]
    const float* __restrict__ b2,              // [E][D]
    const float* __restrict__ x,               // [BS][D]
    const int*   __restrict__ cnt,
    const int*   __restrict__ lists,
    const float* __restrict__ gates,
    float* __restrict__ contrib)               // [BS*2][D]
{
  __shared__ unsigned short xs[TILE_M][Dm + XPAD];
  __shared__ unsigned short hs[TILE_M][FBLK + HPAD];
  __shared__ int   sl[TILE_M];
  __shared__ int   tk[TILE_M];
  __shared__ float gt[TILE_M];
  __shared__ int   vd[TILE_M];

  const int e    = blockIdx.x >> 7;        // MAX_TILES=128 tiles per expert
  const int tile = blockIdx.x & (MAX_TILES - 1);
  const int ne   = cnt[e];
  if (tile * TILE_M >= ne) return;

  const int tid = threadIdx.x;
  if (tid < TILE_M) {
    int idx   = tile * TILE_M + tid;
    int valid = (idx < ne);
    int src   = valid ? idx : 0;
    int entry = lists[e * BSm + src];
    sl[tid] = entry;
    tk[tid] = entry >> 1;
    gt[tid] = valid ? gates[e * BSm + idx] : 0.0f;
    vd[tid] = valid;
  }
  __syncthreads();

  // x tile -> bf16 LDS
  for (int i = tid; i < TILE_M * Dm; i += 256) {
    int m = i >> 10, d = i & (Dm - 1);
    xs[m][d] = f2bf(x[(size_t)tk[m] * Dm + d]);
  }
  __syncthreads();

  const int wave = tid >> 5;
  const int lane = tid & 31;
  const int hi   = lane >> 4;     // half-wave select (K interleave)
  const int ln   = lane & 15;     // M for A-frags, N for B/C-frags

  v8f accD[2][8];
  #pragma unroll
  for (int s = 0; s < 2; s++)
    #pragma unroll
    for (int t = 0; t < 8; t++)
      accD[s][t] = (v8f){0.f,0.f,0.f,0.f,0.f,0.f,0.f,0.f};

  const int d0 = wave * 128;
  const unsigned short* w2base = w2bf + ((size_t)e * Dm + d0) * Fm;
  const unsigned short* xrow0  = &xs[ln][0];
  const unsigned short* xrow1  = &xs[16 + ln][0];

  for (int fb = 0; fb < Fm; fb += FBLK) {
    // ---- Phase A: wave computes h[:, fb+wave*16 .. +15] for 32 rows
    v8f acc0 = (v8f){0.f,0.f,0.f,0.f,0.f,0.f,0.f,0.f};
    v8f acc1 = (v8f){0.f,0.f,0.f,0.f,0.f,0.f,0.f,0.f};
    const int fcol = fb + wave * 16 + ln;
    const unsigned short* w1col = w1bf + ((size_t)e * Fm + fcol) * Dm;
    #pragma unroll 2
    for (int kb = 0; kb < Dm; kb += 32) {
      __builtin_prefetch(w1col + kb + 128, 0, 0);      // global_prefetch_b8
      Frag a0, a1, b;
      b.u8[0]  = *(const ushort8*)(w1col + kb +      8 * hi);
      b.u8[1]  = *(const ushort8*)(w1col + kb + 16 + 8 * hi);
      a0.u8[0] = *(const ushort8*)(xrow0 + kb +      8 * hi);
      a0.u8[1] = *(const ushort8*)(xrow0 + kb + 16 + 8 * hi);
      a1.u8[0] = *(const ushort8*)(xrow1 + kb +      8 * hi);
      a1.u8[1] = *(const ushort8*)(xrow1 + kb + 16 + 8 * hi);
      acc0 = __builtin_amdgcn_wmma_f32_16x16x32_bf16(
               false, a0.v, false, b.v, (short)0, acc0, false, false);
      acc1 = __builtin_amdgcn_wmma_f32_16x16x32_bf16(
               false, a1.v, false, b.v, (short)0, acc1, false, false);
    }
    // bias + exact GELU, stage to LDS as bf16
    float bias = b1[e * Fm + fcol];
    #pragma unroll
    for (int r = 0; r < 8; r++) {
      float h0 = acc0[r] + bias;
      h0 = 0.5f * h0 * (1.0f + erff(h0 * 0.70710678118f));
      hs[r + 8 * hi][wave * 16 + ln] = f2bf(h0);
      float h1 = acc1[r] + bias;
      h1 = 0.5f * h1 * (1.0f + erff(h1 * 0.70710678118f));
      hs[16 + r + 8 * hi][wave * 16 + ln] = f2bf(h1);
    }
    __syncthreads();

    // ---- Phase B: accD += h_blk @ w2[fb:fb+128, d0:d0+128], 32 rows
    const unsigned short* hrow0 = &hs[ln][0];
    const unsigned short* hrow1 = &hs[16 + ln][0];
    #pragma unroll
    for (int kk = 0; kk < FBLK; kk += 32) {
      Frag a0, a1;
      a0.u8[0] = *(const ushort8*)(hrow0 + kk +      8 * hi);
      a0.u8[1] = *(const ushort8*)(hrow0 + kk + 16 + 8 * hi);
      a1.u8[0] = *(const ushort8*)(hrow1 + kk +      8 * hi);
      a1.u8[1] = *(const ushort8*)(hrow1 + kk + 16 + 8 * hi);
      #pragma unroll
      for (int t = 0; t < 8; t++) {
        Frag b;
        const unsigned short* w2col =
            w2base + (size_t)(t * 16 + ln) * Fm + fb + kk;
        b.u8[0] = *(const ushort8*)(w2col +      8 * hi);
        b.u8[1] = *(const ushort8*)(w2col + 16 + 8 * hi);
        accD[0][t] = __builtin_amdgcn_wmma_f32_16x16x32_bf16(
                       false, a0.v, false, b.v, (short)0, accD[0][t], false, false);
        accD[1][t] = __builtin_amdgcn_wmma_f32_16x16x32_bf16(
                       false, a1.v, false, b.v, (short)0, accD[1][t], false, false);
      }
    }
    __syncthreads();
  }

  // ---- store: gate * (acc + b2) into exclusive per-(token,k) slot
  #pragma unroll
  for (int t = 0; t < 8; t++) {
    int d = d0 + t * 16 + ln;
    float bias2 = b2[e * Dm + d];
    #pragma unroll
    for (int s = 0; s < 2; s++) {
      #pragma unroll
      for (int r = 0; r < 8; r++) {
        int m = s * 16 + r + 8 * hi;
        if (vd[m]) {
          float val = gt[m] * (accD[s][t][r] + bias2);
          contrib[(size_t)sl[m] * Dm + d] = val;
        }
      }
    }
  }
}

// ------------------------------------------------------------------- combine
__global__ void moe_combine(const float* __restrict__ x,
                            const float* __restrict__ contrib,
                            float* __restrict__ out) {
  size_t i = (size_t)blockIdx.x * blockDim.x + threadIdx.x;
  if (i >= (size_t)BSm * Dm) return;
  size_t t = i / Dm;
  size_t d = i - t * Dm;
  out[i] = x[i] + contrib[(t * 2 + 0) * Dm + d] + contrib[(t * 2 + 1) * Dm + d];
}

// ===================================================================
extern "C" void kernel_launch(void* const* d_in, const int* in_sizes, int n_in,
                              void* d_out, int out_size, void* d_ws, size_t ws_size,
                              hipStream_t stream) {
  const float* x  = (const float*)d_in[0];   // [B,S,D]
  const float* rw = (const float*)d_in[1];   // [D,E]
  const float* rb = (const float*)d_in[2];   // [E]
  const float* w1 = (const float*)d_in[3];   // [E,D,F]
  const float* b1 = (const float*)d_in[4];   // [E,F]
  const float* w2 = (const float*)d_in[5];   // [E,F,D]
  const float* b2 = (const float*)d_in[6];   // [E,D]
  float* out = (float*)d_out;

  // workspace layout
  char* ws = (char*)d_ws;
  const size_t wElems = (size_t)Em * Dm * Fm;             // 33,554,432
  unsigned short* w1bf    = (unsigned short*)(ws);                        // 64 MiB
  unsigned short* w2bf    = (unsigned short*)(ws + wElems * 2);           // 64 MiB
  float*          contrib = (float*)(ws + wElems * 4);                    // 32 MiB
  char* p = ws + wElems * 4 + (size_t)BSm * 2 * Dm * 4;
  int*   cnt   = (int*)p;            p += 256;
  int*   lists = (int*)p;            p += (size_t)Em * BSm * 4;
  float* gates = (float*)p;

  moe_zero_cnt<<<1, 32, 0, stream>>>(cnt);

  {
    int blocks = (int)((wElems + 255) / 256);
    moe_conv_w1<<<blocks, 256, 0, stream>>>(w1, w1bf);
    moe_conv_w2<<<blocks, 256, 0, stream>>>(w2, w2bf);
  }

  moe_router<<<BSm / 8, 256, 0, stream>>>(x, rw, rb, cnt, lists, gates);

  moe_ffn<<<Em * MAX_TILES, 256, 0, stream>>>(w1bf, w2bf, b1, b2, x,
                                              cnt, lists, gates, contrib);

  {
    int total = BSm * Dm;
    moe_combine<<<(total + 255) / 256, 256, 0, stream>>>(x, contrib, out);
  }
}
